// HM_LSTM_13778255086022
// MI455X (gfx1250) — matrix operand order repaired and checked by reading, not compiled, not run on hardware
//
#include <hip/hip_runtime.h>

#define TT   2048
#define NB   64
#define DD   256
#define HH   512
#define NWG  16
#define NTHREADS 256

typedef __attribute__((ext_vector_type(16))) _Float16 v16h;
typedef __attribute__((ext_vector_type(8)))  _Float16 v8h;
typedef __attribute__((ext_vector_type(8)))  float    v8f;
typedef unsigned int u32x4 __attribute__((ext_vector_type(4)));
typedef int          i32x8 __attribute__((ext_vector_type(8)));
typedef int          i32x4 __attribute__((ext_vector_type(4)));

// ---- output layout (floats): h1s, h2s, z1s, z2s, carry(h1,c1,z1,h2,c2,z2)
#define OFF_H2S ((size_t)NB * TT * HH)
#define OFF_Z1S (2 * OFF_H2S)
#define OFF_Z2S (OFF_Z1S + (size_t)NB * TT)
#define OFF_CH1 (OFF_Z2S + (size_t)NB * TT)
#define OFF_CC1 (OFF_CH1 + (size_t)HH * NB)
#define OFF_CZ1 (OFF_CC1 + (size_t)HH * NB)
#define OFF_CH2 (OFF_CZ1 + (size_t)NB)
#define OFF_CC2 (OFF_CH2 + (size_t)HH * NB)
#define OFF_CZ2 (OFF_CC2 + (size_t)HH * NB)

// ---- dynamic LDS layout (bytes)
#define FSL_OFF   0        // 32 KB: gates [4][32][64] f32
#define ZRED_OFF  32768    //  2 KB: z-row reduction
#define XLS_OFF   34816    // 32 KB: x_t  f16 [64][256]
#define H1LS_OFF  67584    // 64 KB: h1   f16 [64][512]
#define H2LS_OFF  133120   // 64 KB: h2   f16 [64][512]
#define SMEM_BYTES 198656

#define WMMA(a, b, c) \
  __builtin_amdgcn_wmma_f32_16x16x32_f16(false, (a), false, (b), (short)0, (c), false, false)

__device__ __forceinline__ float sigm(float x) { return 1.0f / (1.0f + __expf(-x)); }

// A fragment: 16(M) x 32(K) f16, row-major weights in global, ld=K.
__device__ __forceinline__ v16h fragA(const _Float16* __restrict__ W, int ldk, int row0,
                                      int kb, int lane) {
  int r  = lane & 15;
  int ko = (lane >> 4) << 3;
  const _Float16* p = W + (size_t)(row0 + r) * ldk + kb + ko;
  v8h lo = *(const v8h*)(p);
  v8h hi = *(const v8h*)(p + 16);
  return __builtin_shufflevector(lo, hi, 0, 1, 2, 3, 4, 5, 6, 7, 8, 9, 10, 11, 12, 13, 14, 15);
}

// B fragment: 32(K) x 16(N) f16 from LDS; source stored [batch][feat] (N x K), ld=K.
__device__ __forceinline__ v16h fragB(const _Float16* Hm, int ldk, int col0, int kb, int lane) {
  int n  = lane & 15;
  int ko = (lane >> 4) << 3;
  const _Float16* p = Hm + (size_t)(col0 + n) * ldk + kb + ko;
  v8h lo = *(const v8h*)(p);
  v8h hi = *(const v8h*)(p + 16);
  return __builtin_shufflevector(lo, hi, 0, 1, 2, 3, 4, 5, 6, 7, 8, 9, 10, 11, 12, 13, 14, 15);
}

// ---- Tensor Data Mover: 2-D contiguous tile (rows x kElems, 2-byte elements) -> LDS
// D# per CDNA5 ISA: count=1, type=2, data_size=1 (2B),
// tensor_dim0 = tile_dim0 = kElems, tensor_dim1 = tile_dim1 = rows, stride0 = kElems.
__device__ __forceinline__ void tdm_load_2d(unsigned lds_addr, const _Float16* gsrc,
                                            unsigned kElems, unsigned rows) {
  unsigned long long ga = (unsigned long long)(uintptr_t)gsrc;
  u32x4 g0;
  g0[0] = 1u;                                                   // count=1, user mode
  g0[1] = lds_addr;                                             // lds_addr (bytes)
  g0[2] = (unsigned)ga;                                         // global_addr[31:0]
  g0[3] = (unsigned)((ga >> 32) & 0x01FFFFFFull) | (2u << 30);  // addr[56:32] | type=2
  i32x8 g1;
  g1[0] = (int)(1u << 16);                                   // data_size=2B, no flags
  g1[1] = (int)((kElems & 0xFFFFu) << 16);                   // tensor_dim0[15:0]
  g1[2] = (int)((kElems >> 16) | ((rows & 0xFFFFu) << 16));  // td0[31:16] | td1[15:0]
  g1[3] = (int)((rows >> 16) | ((kElems & 0xFFFFu) << 16));  // td1[31:16] | tile_dim0
  g1[4] = (int)(rows & 0xFFFFu);                             // tile_dim1 | tile_dim2=0
  g1[5] = (int)kElems;                                       // tensor_dim0_stride[31:0]
  g1[6] = 0;
  g1[7] = 0;
  i32x4 z4 = {0, 0, 0, 0};
#if __clang_major__ >= 23
  i32x8 z8v = {0, 0, 0, 0, 0, 0, 0, 0};
  __builtin_amdgcn_tensor_load_to_lds(g0, g1, z4, z4, z8v, 0);
#else
  __builtin_amdgcn_tensor_load_to_lds(g0, g1, z4, z4, 0);
#endif
}

__device__ void gbarrier(int* bar) {
  __syncthreads();
  if (threadIdx.x == 0) {
    __threadfence();
    int gen  = __hip_atomic_load(&bar[1], __ATOMIC_RELAXED, __HIP_MEMORY_SCOPE_AGENT);
    int prev = __hip_atomic_fetch_add(&bar[0], 1, __ATOMIC_ACQ_REL, __HIP_MEMORY_SCOPE_AGENT);
    if (prev == NWG - 1) {
      __hip_atomic_store(&bar[0], 0, __ATOMIC_RELAXED, __HIP_MEMORY_SCOPE_AGENT);
      __hip_atomic_fetch_add(&bar[1], 1, __ATOMIC_RELEASE, __HIP_MEMORY_SCOPE_AGENT);
    } else {
      while (__hip_atomic_load(&bar[1], __ATOMIC_ACQUIRE, __HIP_MEMORY_SCOPE_AGENT) == gen)
        __builtin_amdgcn_s_sleep(2);
    }
    __threadfence();
  }
  __syncthreads();
}

// ---- f32 -> f16 conversion (weights + inputs; stays L2-resident across recurrence)
__global__ void hm_cvt(const float* __restrict__ src, _Float16* __restrict__ dst, int n) {
  for (int i = blockIdx.x * blockDim.x + threadIdx.x; i < n; i += gridDim.x * blockDim.x)
    dst[i] = (_Float16)src[i];
}

// ---- zero recurrent state + barrier words each launch (graph-replay safe)
__global__ void hm_init(unsigned int* __restrict__ state, int nwords) {
  for (int i = blockIdx.x * blockDim.x + threadIdx.x; i < nwords; i += gridDim.x * blockDim.x)
    state[i] = 0u;
}

__global__ __launch_bounds__(NTHREADS, 1) void hm_main(
    const _Float16* __restrict__ xb, const float* __restrict__ bias1,
    const float* __restrict__ bias2,
    const _Float16* __restrict__ Wb1, const _Float16* __restrict__ U11,
    const _Float16* __restrict__ U21, const _Float16* __restrict__ W02,
    const _Float16* __restrict__ U12,
    _Float16* __restrict__ h1b, _Float16* __restrict__ h2b,
    float* __restrict__ z1b, float* __restrict__ z2b, int* __restrict__ bar,
    float* __restrict__ out) {
  extern __shared__ char smem[];
  float*    fsl  = (float*)(smem + FSL_OFF);
  float*    zred = (float*)(smem + ZRED_OFF);
  _Float16* xls  = (_Float16*)(smem + XLS_OFF);
  _Float16* h1ls = (_Float16*)(smem + H1LS_OFF);
  _Float16* h2ls = (_Float16*)(smem + H2LS_OFF);

  const unsigned xls_a  = (unsigned)(uintptr_t)xls;   // LDS byte addresses for TDM
  const unsigned h1ls_a = (unsigned)(uintptr_t)h1ls;
  const unsigned h2ls_a = (unsigned)(uintptr_t)h2ls;

  const int wg   = blockIdx.x;
  const int tid  = threadIdx.x;
  const int lane = tid & 31;
  const int wave = tid >> 5;
  const int g    = wave >> 1;                 // gate block 0..3
  const int s    = wave & 1;                  // sub-tile
  const int fbase = wg * 32;                  // features owned by this WG
  const int mrow = g * HH + fbase + s * 16;   // gate-matrix row tile

  const v8f z8 = {0.f, 0.f, 0.f, 0.f, 0.f, 0.f, 0.f, 0.f};

  float c1v[8], c2v[8], h2v[8];
#pragma unroll
  for (int k = 0; k < 8; ++k) { c1v[k] = 0.f; c2v[k] = 0.f; h2v[k] = 0.f; }

  for (int t = 0; t < TT; ++t) {
    const int prev = t & 1, cur = prev ^ 1;
    const _Float16* h1p = h1b + (size_t)prev * NB * HH;
    const _Float16* h2p = h2b + (size_t)prev * NB * HH;
    _Float16* h1c = h1b + (size_t)cur * NB * HH;
    _Float16* h2c = h2b + (size_t)cur * NB * HH;
    const float* z1p = z1b + prev * NB;
    float* z1c = z1b + cur * NB;
    const float* z2p = z2b + prev * NB;
    float* z2c = z2b + cur * NB;

    // ---- stage x_t, h1(t-1), h2(t-1) into LDS via Tensor Data Mover (wave 0 issues)
    if (wave == 0) {
      tdm_load_2d(xls_a, xb + (size_t)t * NB * DD, DD, NB);
      tdm_load_2d(h1ls_a, h1p, HH, NB);
      tdm_load_2d(h2ls_a, h2p, HH, NB);
      __builtin_amdgcn_s_wait_tensorcnt(0);
    }
    __syncthreads();

    // ================= layer 1: f_s = W01@x + U11@h1 + z1_prev*(U21@h2) + b ================
    {
      v8f accA[4], accB[4];
#pragma unroll
      for (int q = 0; q < 4; ++q) { accA[q] = z8; accB[q] = z8; }

#pragma unroll 2
      for (int kb = 0; kb < DD; kb += 32) {          // x segment, K=256
        v16h a = fragA(Wb1, DD, mrow, kb, lane);
#pragma unroll
        for (int q = 0; q < 4; ++q) {
          v16h b = fragB(xls, DD, q * 16, kb, lane);
          accA[q] = WMMA(a, b, accA[q]);
        }
      }
#pragma unroll 2
      for (int kb = 0; kb < HH; kb += 32) {          // U11 @ h1_prev, K=512
        v16h a = fragA(U11, HH, mrow, kb, lane);
#pragma unroll
        for (int q = 0; q < 4; ++q) {
          v16h b = fragB(h1ls, HH, q * 16, kb, lane);
          accA[q] = WMMA(a, b, accA[q]);
        }
      }
#pragma unroll 2
      for (int kb = 0; kb < HH; kb += 32) {          // U21 @ h2_prev, K=512 (z1-scaled later)
        v16h a = fragA(U21, HH, mrow, kb, lane);
#pragma unroll
        for (int q = 0; q < 4; ++q) {
          v16h b = fragB(h2ls, HH, q * 16, kb, lane);
          accB[q] = WMMA(a, b, accB[q]);
        }
      }
      const int rlo = lane & 15, mhi = (lane >> 4) * 8;
#pragma unroll
      for (int q = 0; q < 4; ++q) {
        int col = q * 16 + rlo;
        float zc = z1p[col];
#pragma unroll
        for (int rr = 0; rr < 8; ++rr) {
          int Ml = mhi + rr;
          float v = accA[q][rr] + zc * accB[q][rr] + bias1[mrow + Ml];
          fsl[(g * 32 + s * 16 + Ml) * 64 + col] = v;
        }
      }
    }
    if (wg == 0) {  // z-row (row 4H) GEMV partials, reading staged LDS operands
      int n = tid & 63, seg = tid >> 6;
      float pa = 0.f, pt = 0.f;
      const _Float16* wz = Wb1 + (size_t)2048 * DD;
      const _Float16* uz = U11 + (size_t)2048 * HH;
      const _Float16* vz = U21 + (size_t)2048 * HH;
      const _Float16* xr  = xls + (size_t)n * DD;
      const _Float16* h1r = h1ls + (size_t)n * HH;
      const _Float16* h2r = h2ls + (size_t)n * HH;
      for (int k = seg * 64; k < seg * 64 + 64; ++k) pa += (float)wz[k] * (float)xr[k];
      for (int k = seg * 128; k < seg * 128 + 128; ++k) {
        pa += (float)uz[k] * (float)h1r[k];
        pt += (float)vz[k] * (float)h2r[k];
      }
      zred[seg * 64 + n] = pa;
      zred[256 + seg * 64 + n] = pt;
    }
    __syncthreads();
    if (wg == 0 && tid < NB) {
      float a = zred[tid] + zred[64 + tid] + zred[128 + tid] + zred[192 + tid] + bias1[4 * HH];
      float tt2 = zred[256 + tid] + zred[320 + tid] + zred[384 + tid] + zred[448 + tid];
      a += z1p[tid] * tt2;
      float zh = fminf(fmaxf((a + 1.0f) * 0.5f, 0.f), 1.f);  // hard_sigm, A=1
      float zn = zh > 0.5f ? 1.f : 0.f;                      // STE forward
      z1c[tid] = zn;
      out[OFF_Z1S + (size_t)tid * TT + t] = zn;
      if (t == TT - 1) out[OFF_CZ1 + tid] = zn;
    }
    {  // layer-1 cell update (z_bottom == 1 simplification)
      int fl = tid >> 3, n0 = (tid & 7) * 8, r = fbase + fl;
#pragma unroll
      for (int k = 0; k < 8; ++k) {
        int n = n0 + k;
        float z  = z1p[n];
        float fg = sigm(fsl[(0 * 32 + fl) * 64 + n]);
        float ig = sigm(fsl[(1 * 32 + fl) * 64 + n]);
        float og = sigm(fsl[(2 * 32 + fl) * 64 + n]);
        float gg = tanhf(fsl[(3 * 32 + fl) * 64 + n]);
        float iq = ig * gg;
        float cn = iq + (1.f - z) * fg * c1v[k];
        float hn = og * tanhf(cn);
        c1v[k] = cn;
        out[(size_t)(n * TT + t) * HH + r] = hn;
        h1c[(size_t)n * HH + r] = (_Float16)hn;
        if (t == TT - 1) {
          out[OFF_CH1 + (size_t)r * NB + n] = hn;
          out[OFF_CC1 + (size_t)r * NB + n] = cn;
        }
      }
    }
    gbarrier(bar);  // publish h1(t), z1(t)

    // ---- stage h1(t) into LDS (h2(t-1) tile still valid from layer 1)
    if (wave == 0) {
      tdm_load_2d(h1ls_a, h1c, HH, NB);
      __builtin_amdgcn_s_wait_tensorcnt(0);
    }
    __syncthreads();

    // ================= layer 2: f_s = W01_2@h1_new + z1_new*(U11_2@h2_prev) + b =============
    {
      v8f accA[4], accB[4];
#pragma unroll
      for (int q = 0; q < 4; ++q) { accA[q] = z8; accB[q] = z8; }
#pragma unroll 2
      for (int kb = 0; kb < HH; kb += 32) {
        v16h a = fragA(W02, HH, mrow, kb, lane);
#pragma unroll
        for (int q = 0; q < 4; ++q) {
          v16h b = fragB(h1ls, HH, q * 16, kb, lane);
          accA[q] = WMMA(a, b, accA[q]);
        }
      }
#pragma unroll 2
      for (int kb = 0; kb < HH; kb += 32) {
        v16h a = fragA(U12, HH, mrow, kb, lane);
#pragma unroll
        for (int q = 0; q < 4; ++q) {
          v16h b = fragB(h2ls, HH, q * 16, kb, lane);
          accB[q] = WMMA(a, b, accB[q]);
        }
      }
      const int rlo = lane & 15, mhi = (lane >> 4) * 8;
#pragma unroll
      for (int q = 0; q < 4; ++q) {
        int col = q * 16 + rlo;
        float zc = z1c[col];
#pragma unroll
        for (int rr = 0; rr < 8; ++rr) {
          int Ml = mhi + rr;
          float v = accA[q][rr] + zc * accB[q][rr] + bias2[mrow + Ml];
          fsl[(g * 32 + s * 16 + Ml) * 64 + col] = v;
        }
      }
    }
    if (wg == 0) {  // layer-2 z-row partials
      int n = tid & 63, seg = tid >> 6;
      float pa = 0.f, pt = 0.f;
      const _Float16* wz = W02 + (size_t)2048 * HH;
      const _Float16* uz = U12 + (size_t)2048 * HH;
      const _Float16* ha = h1ls + (size_t)n * HH;
      const _Float16* hb = h2ls + (size_t)n * HH;
      for (int k = seg * 128; k < seg * 128 + 128; ++k) {
        pa += (float)wz[k] * (float)ha[k];
        pt += (float)uz[k] * (float)hb[k];
      }
      zred[seg * 64 + n] = pa;
      zred[256 + seg * 64 + n] = pt;
    }
    __syncthreads();
    if (wg == 0 && tid < NB) {
      float a = zred[tid] + zred[64 + tid] + zred[128 + tid] + zred[192 + tid] + bias2[4 * HH];
      float tt2 = zred[256 + tid] + zred[320 + tid] + zred[384 + tid] + zred[448 + tid];
      a += z1c[tid] * tt2;
      float zh = fminf(fmaxf((a + 1.0f) * 0.5f, 0.f), 1.f);
      float zn = zh > 0.5f ? 1.f : 0.f;
      z2c[tid] = zn;
      out[OFF_Z2S + (size_t)tid * TT + t] = zn;
      if (t == TT - 1) out[OFF_CZ2 + tid] = zn;
    }
    {  // layer-2 cell update (full HM-LSTM mixing)
      int fl = tid >> 3, n0 = (tid & 7) * 8, r = fbase + fl;
#pragma unroll
      for (int k = 0; k < 8; ++k) {
        int n = n0 + k;
        float z  = z2p[n];
        float zb = z1c[n];
        float fg = sigm(fsl[(0 * 32 + fl) * 64 + n]);
        float ig = sigm(fsl[(1 * 32 + fl) * 64 + n]);
        float og = sigm(fsl[(2 * 32 + fl) * 64 + n]);
        float gg = tanhf(fsl[(3 * 32 + fl) * 64 + n]);
        float iq = ig * gg;
        float cn = z * iq + (1.f - z) * (1.f - zb) * c2v[k] + (1.f - z) * zb * (fg * c2v[k] + iq);
        float th = tanhf(cn);
        float hn = z * og * th + (1.f - z) * (1.f - zb) * h2v[k] + (1.f - z) * zb * og * th;
        c2v[k] = cn;
        h2v[k] = hn;
        out[OFF_H2S + (size_t)(n * TT + t) * HH + r] = hn;
        h2c[(size_t)n * HH + r] = (_Float16)hn;
        if (t == TT - 1) {
          out[OFF_CH2 + (size_t)r * NB + n] = hn;
          out[OFF_CC2 + (size_t)r * NB + n] = cn;
        }
      }
    }
    gbarrier(bar);  // publish h2(t), z2(t)
  }
}

extern "C" void kernel_launch(void* const* d_in, const int* in_sizes, int n_in,
                              void* d_out, int out_size, void* d_ws, size_t ws_size,
                              hipStream_t stream) {
  (void)in_sizes; (void)n_in; (void)out_size; (void)ws_size;
  // input order: inputs, U11_1, U21_1, W01_1, bias_1, U11_2, W01_2, bias_2
  const float* x    = (const float*)d_in[0];
  const float* U11f = (const float*)d_in[1];
  const float* U21f = (const float*)d_in[2];
  const float* W01f = (const float*)d_in[3];
  const float* b1   = (const float*)d_in[4];
  const float* U12f = (const float*)d_in[5];
  const float* W02f = (const float*)d_in[6];
  const float* b2   = (const float*)d_in[7];
  float* out = (float*)d_out;

  // workspace (~77 MB): f16 weights + f16 inputs + double-buffered f16 h-state + z + barrier
  const size_t N_WB1 = (size_t)2049 * 256;
  const size_t N_U   = (size_t)2049 * 512;
  const size_t N_X   = (size_t)TT * NB * DD;
  const size_t N_HB  = (size_t)2 * NB * HH;  // two parities
  _Float16* Wb1 = (_Float16*)d_ws;
  _Float16* U11 = Wb1 + N_WB1;
  _Float16* U21 = U11 + N_U;
  _Float16* W02 = U21 + N_U;
  _Float16* U12 = W02 + N_U;
  _Float16* xb  = U12 + N_U;
  _Float16* h1b = xb + N_X;
  _Float16* h2b = h1b + N_HB;
  float* z1b = (float*)(h2b + N_HB);
  float* z2b = z1b + 2 * NB;
  int* bar   = (int*)(z2b + 2 * NB);

  int state_words = (int)((2 * N_HB * sizeof(_Float16) + 4 * NB * sizeof(float) + 64) / 4);
  hm_init<<<64, 256, 0, stream>>>((unsigned int*)h1b, state_words);

  hm_cvt<<<1024, 256, 0, stream>>>(W01f, Wb1, (int)N_WB1);
  hm_cvt<<<1024, 256, 0, stream>>>(U11f, U11, (int)N_U);
  hm_cvt<<<1024, 256, 0, stream>>>(U21f, U21, (int)N_U);
  hm_cvt<<<1024, 256, 0, stream>>>(W02f, W02, (int)N_U);
  hm_cvt<<<1024, 256, 0, stream>>>(U12f, U12, (int)N_U);
  hm_cvt<<<4096, 256, 0, stream>>>(x, xb, (int)N_X);

  hm_main<<<NWG, NTHREADS, SMEM_BYTES, stream>>>(xb, b1, b2, Wb1, U11, U21, W02, U12,
                                                 h1b, h2b, z1b, z2b, bar, out);
}
